// Attention_27350351741194
// MI455X (gfx1250) — matrix operand order repaired and checked by reading, not compiled, not run on hardware
//
#include <hip/hip_runtime.h>
#include <hip/hip_bf16.h>
#include <stdint.h>

// ---------------------------------------------------------------------------
// Problem constants (from reference): B=2, N=2048, D=1024, H=16, DH=64
// ---------------------------------------------------------------------------
#define BATCH 2
#define SEQ   2048
#define DMODEL 1024
#define HEADS 16
#define DHEAD 64
#define INNER 1024          // HEADS*DHEAD
#define QKVC  3072          // INNER + 2*INNER
#define MTOK  (BATCH*SEQ)   // 4096 token rows

// WMMA fragment/vector types (CDNA5 wave32)
typedef __bf16          v16bf  __attribute__((ext_vector_type(16)));
typedef float           v8f    __attribute__((ext_vector_type(8)));
typedef unsigned short  ushort8  __attribute__((ext_vector_type(8)));
typedef unsigned short  ushort16 __attribute__((ext_vector_type(16)));
typedef unsigned int    v4u __attribute__((ext_vector_type(4)));
typedef int             v8i __attribute__((ext_vector_type(8)));
typedef int             v4i __attribute__((ext_vector_type(4)));

#if defined(__has_builtin)
#if __has_builtin(__builtin_amdgcn_tensor_load_to_lds)
#define HAVE_TDM 1
#endif
#endif

// float -> bf16 (round-to-nearest-even), stored as raw u16
static __device__ __forceinline__ unsigned short f2bf(float f) {
  unsigned u = __builtin_bit_cast(unsigned, f);
  u += 0x7FFFu + ((u >> 16) & 1u);
  return (unsigned short)(u >> 16);
}

// Build a v16bf fragment out of two 16B LDS/global chunks
static __device__ __forceinline__ v16bf load_frag(const unsigned short* p0,
                                                  const unsigned short* p1) {
  ushort8 lo = *(const ushort8*)p0;
  ushort8 hi = *(const ushort8*)p1;
  ushort16 w = __builtin_shufflevector(lo, hi, 0,1,2,3,4,5,6,7,8,9,10,11,12,13,14,15);
  return __builtin_bit_cast(v16bf, w);
}

static __device__ __forceinline__ v8f wmma_bf16(v16bf a, v16bf b, v8f c) {
  return __builtin_amdgcn_wmma_f32_16x16x32_bf16(false, a, false, b, (short)0, c,
                                                 false, false);
}

// ---------------------------------------------------------------------------
// Tensor Data Mover: 2D tile (tile_d0 contiguous bf16 elems x tile_d1 rows)
// from global into LDS, with 16B pad per 64B row => padded LDS stride of
// 40 ushorts (bank-conflict-free fragment gathers).
// ---------------------------------------------------------------------------
static __device__ __forceinline__ void tdm_load_2d(unsigned lds_addr,
                                                   const void* gptr,
                                                   unsigned tensor_d0,
                                                   unsigned tensor_d1,
                                                   unsigned long long stride0_elems,
                                                   unsigned tile_d0,
                                                   unsigned tile_d1) {
#if HAVE_TDM
  unsigned long long ga = (unsigned long long)(uintptr_t)gptr;
  v4u g0;
  g0[0] = 1u;                                   // count=1, user descriptor
  g0[1] = lds_addr;                             // lds byte address
  g0[2] = (unsigned)ga;                         // global_addr[31:0]
  g0[3] = (unsigned)((ga >> 32) & 0x1FFFFFFu) | (2u << 30); // addr[56:32], type=2
  v8i g1;
  // data_size=1 (2B) | pad_enable | pad_interval=3 (16 DW=64B) | pad_amount=3 (4 DW=16B)
  g1[0] = (int)((1u << 16) | (1u << 20) | (3u << 22) | (3u << 25));
  g1[1] = (int)((tensor_d0 & 0xFFFFu) << 16);                     // dim0[15:0]
  g1[2] = (int)((tensor_d0 >> 16) | ((tensor_d1 & 0xFFFFu) << 16));
  g1[3] = (int)((tensor_d1 >> 16) | (tile_d0 << 16));
  g1[4] = (int)(tile_d1 & 0xFFFFu);             // tile_dim1 | tile_dim2=0
  g1[5] = (int)(unsigned)stride0_elems;         // tensor_dim0_stride[31:0]
  g1[6] = (int)((unsigned)(stride0_elems >> 32) & 0xFFFFu); // stride0[47:32], stride1=0
  g1[7] = 0;
  v4i z4 = {0, 0, 0, 0};
#if __clang_major__ >= 23
  v8i z8 = {0, 0, 0, 0, 0, 0, 0, 0};
  __builtin_amdgcn_tensor_load_to_lds(g0, g1, z4, z4, z8, 0);
#else
  __builtin_amdgcn_tensor_load_to_lds(g0, g1, z4, z4, 0);
#endif
#else
  (void)lds_addr; (void)gptr; (void)tensor_d0; (void)tensor_d1;
  (void)stride0_elems; (void)tile_d0; (void)tile_d1;
#endif
}

// ---------------------------------------------------------------------------
// Conversion kernels
// ---------------------------------------------------------------------------
__global__ __launch_bounds__(256) void cvt_x_kernel(const float* __restrict__ src,
                                                    unsigned short* __restrict__ dst,
                                                    int n) {
  int i = blockIdx.x * 256 + threadIdx.x;
  if (i < n) dst[i] = f2bf(src[i]);
}

// WqkvT[n][k] = (n<1024 ? Wq[k][n] : Wkv[k][n-1024]) as bf16, shape [3072][1024]
__global__ __launch_bounds__(256) void build_wqkvT_kernel(const float* __restrict__ Wq,
                                                          const float* __restrict__ Wkv,
                                                          unsigned short* __restrict__ Wt) {
  int i = blockIdx.x * 256 + threadIdx.x;      // 3072*1024 total
  int k = i & 1023;
  int nn = i >> 10;
  float v = (nn < INNER) ? Wq[(size_t)k * INNER + nn]
                         : Wkv[(size_t)k * (2 * INNER) + (nn - INNER)];
  Wt[i] = f2bf(v);
}

// WoT[n][k] = Wo[k][n] as bf16, shape [1024][1024]
__global__ __launch_bounds__(256) void build_woT_kernel(const float* __restrict__ Wo,
                                                        unsigned short* __restrict__ Wt) {
  int i = blockIdx.x * 256 + threadIdx.x;
  int k = i & 1023;
  int nn = i >> 10;
  Wt[i] = f2bf(Wo[(size_t)k * DMODEL + nn]);
}

// ---------------------------------------------------------------------------
// bf16 GEMM: C[M x Nc] (f32) = A[M x K] * Bt[Nc x K]^T (+ bias)
// Block tile 64x256, 8 waves (2x4), wave tile 32x64, K-step 32.
// Double-buffered LDS fed by the Tensor Data Mover: the prefetch for tile
// i+1 is issued before computing tile i, and we wait with tensorcnt<=2 so
// only the *older* pair of TDM ops must drain (DMA overlaps WMMA).
// ---------------------------------------------------------------------------
#define TM 64
#define TN 256
#define TK 32
#define LDSK 40   // padded K stride in ushorts (80B, conflict-free)

__global__ __launch_bounds__(256) void gemm_bf16_kernel(
    const unsigned short* __restrict__ A,
    const unsigned short* __restrict__ Bt,
    const float* __restrict__ bias,
    float* __restrict__ C, int M, int Nc, int K) {
  __shared__ unsigned short As[2][TM * LDSK];
  __shared__ unsigned short Bs[2][TN * LDSK];

  const int tid = threadIdx.x;
  const int wid = tid >> 5;
  const int lane = tid & 31;
  const int lm = lane & 15;      // fragment row/col within 16
  const int lh = lane >> 4;      // lane half selector
  const int m0 = blockIdx.y * TM;
  const int n0 = blockIdx.x * TN;
  const int wm = (wid & 1) * 32; // wave row base inside block tile
  const int wn = (wid >> 1) * 64;// wave col base inside block tile

  v8f acc[2][4];
#pragma unroll
  for (int i = 0; i < 2; ++i)
#pragma unroll
    for (int t = 0; t < 4; ++t) acc[i][t] = (v8f){0, 0, 0, 0, 0, 0, 0, 0};

  auto stage = [&](int buf, int kb) {
#if HAVE_TDM
    if (tid == 0) {
      tdm_load_2d((unsigned)(uintptr_t)&As[buf][0], A + (size_t)m0 * K + kb,
                  TK, TM, (unsigned long long)K, TK, TM);
      tdm_load_2d((unsigned)(uintptr_t)&Bs[buf][0], Bt + (size_t)n0 * K + kb,
                  TK, TN, (unsigned long long)K, TK, TN);
    }
#else
    int ar = tid >> 2, ac = (tid & 3) * 8;
    *(ushort8*)&As[buf][ar * LDSK + ac] =
        *(const ushort8*)&A[(size_t)(m0 + ar) * K + kb + ac];
#pragma unroll
    for (int i = 0; i < 4; ++i) {
      int cid = tid + i * 256;
      int br = cid >> 2, bc = (cid & 3) * 8;
      *(ushort8*)&Bs[buf][br * LDSK + bc] =
          *(const ushort8*)&Bt[(size_t)(n0 + br) * K + kb + bc];
    }
#endif
  };

  stage(0, 0);
  int p = 0;
  for (int kb = 0; kb < K; kb += TK) {
    const int nxt = kb + TK;
    if (nxt < K) stage(p ^ 1, nxt);   // prefetch next tile into other buffer
#if HAVE_TDM
    if (wid == 0) {
      if (nxt < K) __builtin_amdgcn_s_wait_tensorcnt(2); // only older pair drains
      else         __builtin_amdgcn_s_wait_tensorcnt(0);
    }
#endif
    __syncthreads();                  // staged buf[p] now visible to all waves

    const unsigned short* ap0 = &As[p][(wm + lm) * LDSK];
    const unsigned short* ap1 = ap0 + 16 * LDSK;
    v16bf a0 = load_frag(ap0 + lh * 8, ap0 + 16 + lh * 8);
    v16bf a1 = load_frag(ap1 + lh * 8, ap1 + 16 + lh * 8);
#pragma unroll
    for (int t = 0; t < 4; ++t) {
      const unsigned short* bp = &Bs[p][(wn + t * 16 + lm) * LDSK + lh * 8];
      v16bf b = load_frag(bp, bp + 16);
      acc[0][t] = wmma_bf16(a0, b, acc[0][t]);
      acc[1][t] = wmma_bf16(a1, b, acc[1][t]);
    }
    __syncthreads();                  // buf[p] free before it is restaged
    p ^= 1;
  }

#pragma unroll
  for (int i = 0; i < 2; ++i) {
#pragma unroll
    for (int t = 0; t < 4; ++t) {
#pragma unroll
      for (int r = 0; r < 8; ++r) {
        int row = m0 + wm + i * 16 + r + 8 * lh;
        int col = n0 + wn + t * 16 + lm;
        float v = acc[i][t][r];
        if (bias) v += bias[col];
        C[(size_t)row * Nc + col] = v;
      }
    }
  }
}

// ---------------------------------------------------------------------------
// RMSNorm + RoPE + layout. One block per token row.
//   QKV f32 [row][3072] -> Qb/Kb/Vb bf16 [b][h][n][dh]; Q pre-scaled by 1/8.
// ---------------------------------------------------------------------------
__global__ __launch_bounds__(256) void normrope_kernel(
    const float* __restrict__ QKV,
    const float* __restrict__ wq, const float* __restrict__ wk,
    unsigned short* __restrict__ Qb, unsigned short* __restrict__ Kb,
    unsigned short* __restrict__ Vb) {
  const int row = blockIdx.x;          // 0..MTOK-1
  const int b = row >> 11;             // /SEQ
  const int n = row & (SEQ - 1);
  const int tid = threadIdx.x;
  const float* qr = QKV + (size_t)row * QKVC;

  __shared__ float red[256];
  __shared__ float stats[2];

  // sum of squares over q[0:1024]
  float s = 0.f;
  for (int i = tid; i < INNER; i += 256) { float v = qr[i]; s += v * v; }
  red[tid] = s; __syncthreads();
  for (int off = 128; off > 0; off >>= 1) {
    if (tid < off) red[tid] += red[tid + off];
    __syncthreads();
  }
  if (tid == 0) stats[0] = rsqrtf(red[0] * (1.f / INNER) + 1e-6f);
  __syncthreads();
  // sum of squares over k[1024:2048]
  s = 0.f;
  for (int i = tid; i < INNER; i += 256) { float v = qr[INNER + i]; s += v * v; }
  red[tid] = s; __syncthreads();
  for (int off = 128; off > 0; off >>= 1) {
    if (tid < off) red[tid] += red[tid + off];
    __syncthreads();
  }
  if (tid == 0) stats[1] = rsqrtf(red[0] * (1.f / INNER) + 1e-6f);
  __syncthreads();

  const float rq = stats[0], rk = stats[1];
  for (int p = tid; p < INNER / 2; p += 256) {
    int i0 = 2 * p, i1 = i0 + 1;
    int h = i0 >> 6;
    int d = i0 & 63;
    int j = d >> 1;
    // theta = 10000^(-2j/64) = exp(-(j/32)*ln(10000))
    float theta = __expf(-(float)j * (0.03125f * 9.2103403719761836f));
    float cs, sn;
    __sincosf((float)n * theta, &cs, &sn);

    size_t base = (((size_t)(b * HEADS + h) * SEQ) + n) * DHEAD + d;

    float x0 = qr[i0] * rq * wq[i0];
    float x1 = qr[i1] * rq * wq[i1];
    Qb[base]     = f2bf((cs * x0 - sn * x1) * 0.125f);  // fold 1/sqrt(DH)
    Qb[base + 1] = f2bf((sn * x0 + cs * x1) * 0.125f);

    float y0 = qr[INNER + i0] * rk * wk[i0];
    float y1 = qr[INNER + i1] * rk * wk[i1];
    Kb[base]     = f2bf(cs * y0 - sn * y1);
    Kb[base + 1] = f2bf(sn * y0 + cs * y1);

    Vb[base]     = f2bf(qr[2 * INNER + i0]);
    Vb[base + 1] = f2bf(qr[2 * INNER + i1]);
  }
}

// ---------------------------------------------------------------------------
// Flash attention: grid (SEQ/64, H, B), 128 threads = 4 waves, wave = 16 queries.
// 32-key steps: 4 score WMMAs, online softmax, P via per-wave LDS, 4 PV WMMAs.
// Next K/V block prefetched with global_prefetch. Writes bf16 [B*N][INNER].
// ---------------------------------------------------------------------------
__global__ __launch_bounds__(128) void attn_kernel(
    const unsigned short* __restrict__ Qb,
    const unsigned short* __restrict__ Kb,
    const unsigned short* __restrict__ Vb,
    unsigned short* __restrict__ Ob) {
  __shared__ unsigned short Vt[64 * LDSK];       // V^T tile: [dh][key(32)+pad]
  __shared__ unsigned short Ps[4][16 * LDSK];    // per-wave P scratch [m][key+pad]

  const int tid = threadIdx.x;
  const int wid = tid >> 5;
  const int lane = tid & 31;
  const int lm = lane & 15;
  const int lh = lane >> 4;
  const int h = blockIdx.y, b = blockIdx.z;
  const int bh = b * HEADS + h;
  const int q0 = blockIdx.x * 64 + wid * 16;

  // Q fragments (K=64 -> two K=32 fragments), loaded once
  const unsigned short* qrow = Qb + ((size_t)bh * SEQ + q0 + lm) * DHEAD;
  v16bf qa0 = load_frag(qrow + lh * 8,      qrow + 16 + lh * 8);
  v16bf qa1 = load_frag(qrow + 32 + lh * 8, qrow + 48 + lh * 8);

  v8f o0 = (v8f){0,0,0,0,0,0,0,0}, o1 = o0, o2 = o0, o3 = o0;
  float mrow[8], lrow[8];
#pragma unroll
  for (int r = 0; r < 8; ++r) { mrow[r] = -1e30f; lrow[r] = 0.f; }

  for (int kb = 0; kb < SEQ; kb += 32) {
    // prefetch next key/value block (one 128B row per lane)
    if (kb + 32 < SEQ) {
      size_t nrow = ((size_t)bh * SEQ + kb + 32 + lane) * DHEAD;
      __builtin_prefetch(Kb + nrow, 0, 0);
      __builtin_prefetch(Vb + nrow, 0, 0);
    }
    __syncthreads();
    // cooperative transpose-load of V block (32 keys x 64 dh) into Vt[dh][key]
#pragma unroll
    for (int i = 0; i < 2; ++i) {
      int c = tid + i * 128;             // 256 chunks of 8
      int key = c >> 3, dc = (c & 7) * 8;
      ushort8 v = *(const ushort8*)&Vb[((size_t)bh * SEQ + kb + key) * DHEAD + dc];
#pragma unroll
      for (int j = 0; j < 8; ++j) Vt[(dc + j) * LDSK + key] = v[j];
    }
    __syncthreads();

    // scores for two 16-key tiles
    v8f s0 = (v8f){0,0,0,0,0,0,0,0}, s1 = s0;
    {
      const unsigned short* kr = Kb + ((size_t)bh * SEQ + kb + lm) * DHEAD;
      s0 = wmma_bf16(qa0, load_frag(kr + lh * 8,      kr + 16 + lh * 8),  s0);
      s0 = wmma_bf16(qa1, load_frag(kr + 32 + lh * 8, kr + 48 + lh * 8), s0);
    }
    {
      const unsigned short* kr = Kb + ((size_t)bh * SEQ + kb + 16 + lm) * DHEAD;
      s1 = wmma_bf16(qa0, load_frag(kr + lh * 8,      kr + 16 + lh * 8),  s1);
      s1 = wmma_bf16(qa1, load_frag(kr + 32 + lh * 8, kr + 48 + lh * 8), s1);
    }

    // online softmax (row M = r + 8*lh lives in register r across 16 lanes)
#pragma unroll
    for (int r = 0; r < 8; ++r) {
      float mx = fmaxf(s0[r], s1[r]);
#pragma unroll
      for (int off = 8; off > 0; off >>= 1) mx = fmaxf(mx, __shfl_xor(mx, off, 16));
      float mn = fmaxf(mrow[r], mx);
      float sc = __expf(mrow[r] - mn);
      mrow[r] = mn;
      float p0 = __expf(s0[r] - mn);
      float p1 = __expf(s1[r] - mn);
      s0[r] = p0; s1[r] = p1;
      float sum = p0 + p1;
#pragma unroll
      for (int off = 8; off > 0; off >>= 1) sum += __shfl_xor(sum, off, 16);
      lrow[r] = lrow[r] * sc + sum;
      o0[r] *= sc; o1[r] *= sc; o2[r] *= sc; o3[r] *= sc;
    }

    // P (16x32) -> per-wave LDS in A-layout-friendly [m][key]
#pragma unroll
    for (int r = 0; r < 8; ++r) {
      Ps[wid][(r + 8 * lh) * LDSK + lm]      = f2bf(s0[r]);
      Ps[wid][(r + 8 * lh) * LDSK + 16 + lm] = f2bf(s1[r]);
    }
    const unsigned short* pp = &Ps[wid][lm * LDSK];
    v16bf pa = load_frag(pp + lh * 8, pp + 16 + lh * 8);

    // O += P * V  (4 dh tiles)
    {
      const unsigned short* vp0 = &Vt[(0 * 16 + lm) * LDSK + lh * 8];
      const unsigned short* vp1 = &Vt[(1 * 16 + lm) * LDSK + lh * 8];
      const unsigned short* vp2 = &Vt[(2 * 16 + lm) * LDSK + lh * 8];
      const unsigned short* vp3 = &Vt[(3 * 16 + lm) * LDSK + lh * 8];
      o0 = wmma_bf16(pa, load_frag(vp0, vp0 + 16), o0);
      o1 = wmma_bf16(pa, load_frag(vp1, vp1 + 16), o1);
      o2 = wmma_bf16(pa, load_frag(vp2, vp2 + 16), o2);
      o3 = wmma_bf16(pa, load_frag(vp3, vp3 + 16), o3);
    }
  }

  // normalize and write out to [B*N][INNER] bf16
#pragma unroll
  for (int r = 0; r < 8; ++r) {
    float inv = 1.f / lrow[r];
    int row = b * SEQ + q0 + r + 8 * lh;
    size_t base = (size_t)row * INNER + h * DHEAD + lm;
    Ob[base + 0]  = f2bf(o0[r] * inv);
    Ob[base + 16] = f2bf(o1[r] * inv);
    Ob[base + 32] = f2bf(o2[r] * inv);
    Ob[base + 48] = f2bf(o3[r] * inv);
  }
}

// ---------------------------------------------------------------------------
// Launcher
// ---------------------------------------------------------------------------
extern "C" void kernel_launch(void* const* d_in, const int* in_sizes, int n_in,
                              void* d_out, int out_size, void* d_ws, size_t ws_size,
                              hipStream_t stream) {
  (void)in_sizes; (void)n_in; (void)out_size; (void)ws_size;
  const float* x    = (const float*)d_in[0];
  const float* Wq   = (const float*)d_in[1];
  const float* Wkv  = (const float*)d_in[2];
  const float* nqw  = (const float*)d_in[3];
  const float* nkw  = (const float*)d_in[4];
  const float* Wo   = (const float*)d_in[5];
  const float* bo   = (const float*)d_in[6];
  float* out = (float*)d_out;

  char* ws = (char*)d_ws;
  size_t off = 0;
  auto carve = [&](size_t bytes) { char* p = ws + off; off += (bytes + 255) & ~(size_t)255; return p; };
  unsigned short* Xb     = (unsigned short*)carve((size_t)MTOK * DMODEL * 2);   // 8 MB
  unsigned short* WqkvT  = (unsigned short*)carve((size_t)QKVC * DMODEL * 2);   // 6 MB
  unsigned short* WoT    = (unsigned short*)carve((size_t)DMODEL * INNER * 2);  // 2 MB
  float*          QKV    = (float*)carve((size_t)MTOK * QKVC * 4);              // 48 MB
  unsigned short* Qb     = (unsigned short*)carve((size_t)BATCH * HEADS * SEQ * DHEAD * 2);
  unsigned short* Kb     = (unsigned short*)carve((size_t)BATCH * HEADS * SEQ * DHEAD * 2);
  unsigned short* Vb     = (unsigned short*)carve((size_t)BATCH * HEADS * SEQ * DHEAD * 2);
  unsigned short* Attnb  = (unsigned short*)carve((size_t)MTOK * INNER * 2);    // 8 MB

  // 1) conversions
  cvt_x_kernel<<<(MTOK * DMODEL + 255) / 256, 256, 0, stream>>>(x, Xb, MTOK * DMODEL);
  build_wqkvT_kernel<<<(QKVC * DMODEL) / 256, 256, 0, stream>>>(Wq, Wkv, WqkvT);
  build_woT_kernel<<<(DMODEL * INNER) / 256, 256, 0, stream>>>(Wo, WoT);

  // 2) fused QKV projection: [4096 x 3072] = Xb[4096x1024] * WqkvT^T
  gemm_bf16_kernel<<<dim3(QKVC / TN, MTOK / TM), 256, 0, stream>>>(
      Xb, WqkvT, nullptr, QKV, MTOK, QKVC, DMODEL);

  // 3) RMSNorm + RoPE + head-major bf16 layout
  normrope_kernel<<<MTOK, 256, 0, stream>>>(QKV, nqw, nkw, Qb, Kb, Vb);

  // 4) attention
  attn_kernel<<<dim3(SEQ / 64, HEADS, BATCH), 128, 0, stream>>>(Qb, Kb, Vb, Attnb);

  // 5) output projection + bias -> f32 d_out
  gemm_bf16_kernel<<<dim3(DMODEL / TN, MTOK / TM), 256, 0, stream>>>(
      Attnb, WoT, bo, out, MTOK, DMODEL, INNER);
}
